// GAT_42760694399683
// MI455X (gfx1250) — matrix operand order repaired and checked by reading, not compiled, not run on hardware
//
#include <hip/hip_runtime.h>
#include <hip/hip_bf16.h>

typedef __attribute__((ext_vector_type(16))) _Float16 v16h;
typedef __attribute__((ext_vector_type(8)))  float    v8f;

// ---------------- monotonic float<->uint mapping for exact atomic float max ----
static __device__ __forceinline__ unsigned f2mono(float f) {
  unsigned u = __float_as_uint(f);
  return (u & 0x80000000u) ? ~u : (u | 0x80000000u);
}
static __device__ __forceinline__ float mono2f(unsigned u) {
  return (u & 0x80000000u) ? __uint_as_float(u & 0x7fffffffu) : __uint_as_float(~u);
}
#define MONO_NEG_INF 0x007FFFFFu  // f2mono(-inf)

// ---------------- generic 32-bit fill (graph-capture-safe memset) -------------
__global__ void k_fill_u32(unsigned* __restrict__ p, unsigned v, long long n) {
  long long i = (long long)blockIdx.x * blockDim.x + threadIdx.x;
  if (i < n) p[i] = v;
}

// ---------------- WMMA GEMM: H[N,64] = A[N,K] x W[K,64], K in {16,64} --------
// 256-thread block = 8 waves; block covers 32 rows x 64 cols (8 16x16 tiles,
// one per wave). W is staged once per block into LDS, transposed and converted
// to f16, with a zero-filled 16-column pad so the K=16 -> K=32 WMMA zero-pad
// falls out of the pad region with no branches. A-fragment: four
// global_load_b128 per lane (rows contiguous). B-fragment: two ds_load_b128
// per lane. K is a template parameter so everything unrolls branch-free.
template <int K>
__global__ __launch_bounds__(256)
void k_gemm_wmma(const float* __restrict__ A, const float* __restrict__ W,
                 float* __restrict__ H, int N) {
  constexpr int LDW = K + 16;                       // f16 elems per LDS row (+pad)
  __shared__ alignas(32) _Float16 Wt[64 * LDW];

  // cooperative stage: Wt[c][k] = (f16) W[k][c]; pad k in [K, LDW) = 0
  for (int idx = threadIdx.x; idx < 64 * LDW; idx += 256) {
    int c = idx / LDW, k = idx % LDW;
    Wt[idx] = (k < K) ? (_Float16)W[k * 64 + c] : (_Float16)0.0f;
  }
  __syncthreads();

  const int wave = threadIdx.x >> 5;
  const int lane = threadIdx.x & 31;
  const int half = lane >> 4;                       // 0: lanes 0-15, 1: 16-31
  const int l16  = lane & 15;
  const int row0 = blockIdx.x * 32 + (wave >> 2) * 16;
  const int col0 = (wave & 3) * 16;

  int arow = row0 + l16;
  if (arow > N - 1) arow = N - 1;                   // clamp (stores are guarded)
  const float* Arow = A + (long long)arow * K;

  v8f acc = {};
#pragma unroll
  for (int kofs = 0; kofs < K; kofs += 32) {
    // A fragment 16x32 (MxK): lanes 0-15 hold K {0..7,16..23}, lanes 16-31
    // hold K {8..15,24..31} of row l16.
    const int klo = kofs + (half ? 8 : 0);
    const int khi = kofs + (half ? 24 : 16);
    float4 a0 = *(const float4*)(Arow + klo);
    float4 a1 = *(const float4*)(Arow + klo + 4);
    float4 a2, a3;
    if (khi + 8 <= K) {                             // folds at compile time
      a2 = *(const float4*)(Arow + khi);
      a3 = *(const float4*)(Arow + khi + 4);
    } else {
      a2 = make_float4(0.f, 0.f, 0.f, 0.f);
      a3 = make_float4(0.f, 0.f, 0.f, 0.f);
    }
    v16h a;
    a[0]  = (_Float16)a0.x; a[1]  = (_Float16)a0.y;
    a[2]  = (_Float16)a0.z; a[3]  = (_Float16)a0.w;
    a[4]  = (_Float16)a1.x; a[5]  = (_Float16)a1.y;
    a[6]  = (_Float16)a1.z; a[7]  = (_Float16)a1.w;
    a[8]  = (_Float16)a2.x; a[9]  = (_Float16)a2.y;
    a[10] = (_Float16)a2.z; a[11] = (_Float16)a2.w;
    a[12] = (_Float16)a3.x; a[13] = (_Float16)a3.y;
    a[14] = (_Float16)a3.z; a[15] = (_Float16)a3.w;

    // B fragment 32x16 (KxN): lane carries column col0+l16, K {kb..kb+7, kb+16..kb+23}.
    // For K=16, the upper 8 (k>=16) read the zeroed pad -> implicit zero-pad, no branch.
    const int kb = kofs + (half ? 8 : 0);
    typedef __attribute__((ext_vector_type(8))) _Float16 v8h_;
    const _Float16* bp = Wt + (col0 + l16) * LDW + kb;
    v8h_ b0 = *(const v8h_*)(bp);
    v8h_ b1 = *(const v8h_*)(bp + 16);
    v16h b = __builtin_shufflevector(b0, b1, 0, 1, 2, 3, 4, 5, 6, 7, 8, 9, 10, 11, 12, 13, 14, 15);

    acc = __builtin_amdgcn_wmma_f32_16x16x32_f16(
        /*neg_a=*/false, a, /*neg_b=*/false, b,
        /*c_mod=*/(short)0, acc, /*reuse_a=*/false, /*reuse_b=*/false);
  }
  // D layout: VGPR v -> row (v + half*8), col l16.
#pragma unroll
  for (int v = 0; v < 8; ++v) {
    int r = row0 + v + (half << 3);
    if (r < N) H[(long long)r * 64 + col0 + l16] = acc[v];
  }
}

// ---------------- per-(node,head) attention logits ---------------------------
__global__ void k_alpha(const float* __restrict__ H, const float* __restrict__ a_s,
                        const float* __restrict__ a_d, float* __restrict__ als,
                        float* __restrict__ ald, int N) {
  int t = blockIdx.x * blockDim.x + threadIdx.x;
  if (t >= N * 4) return;
  int h = t & 3;
  const float* hp = H + (long long)(t >> 2) * 64 + h * 16;
  const float* sp = a_s + h * 16;
  const float* dp = a_d + h * 16;
  float ss = 0.f, sd = 0.f;
#pragma unroll
  for (int c = 0; c < 16; ++c) { float v = hp[c]; ss += v * sp[c]; sd += v * dp[c]; }
  als[t] = ss;
  ald[t] = sd;
}

// ---------------- segment softmax pass 1: leaky_relu + per-dst max -----------
__global__ void k_edge_max(const int* __restrict__ ei, int E, int N,
                           const float* __restrict__ als, const float* __restrict__ ald,
                           float* __restrict__ ebuf, unsigned* __restrict__ segmax) {
  int t = blockIdx.x * blockDim.x + threadIdx.x;
  if (t >= (E + N) * 4) return;
  int eid = t >> 2, h = t & 3;
  int s, d;
  if (eid < E) { s = ei[eid]; d = ei[E + eid]; } else { s = d = eid - E; }
  float e = als[s * 4 + h] + ald[d * 4 + h];
  e = (e > 0.f) ? e : 0.2f * e;               // leaky_relu, NEG_SLOPE=0.2
  ebuf[t] = e;
  atomicMax(&segmax[d * 4 + h], f2mono(e));   // exact, order-independent
}

// ---------------- pass 2: exp(e - max), per-dst denom ------------------------
__global__ void k_edge_den(const int* __restrict__ ei, int E, int N,
                           const unsigned* __restrict__ segmax,
                           float* __restrict__ ebuf, float* __restrict__ segden) {
  int t = blockIdx.x * blockDim.x + threadIdx.x;
  if (t >= (E + N) * 4) return;
  int eid = t >> 2, h = t & 3;
  int d = (eid < E) ? ei[E + eid] : (eid - E);
  float ex = __expf(ebuf[t] - mono2f(segmax[d * 4 + h]));
  ebuf[t] = ex;
  atomicAdd(&segden[d * 4 + h], ex);
}

// ---------------- pass 3: out[dst] += alpha * h[src] -------------------------
__global__ void k_edge_aggr(const int* __restrict__ ei, int E, int N,
                            const float* __restrict__ H, const float* __restrict__ ebuf,
                            const float* __restrict__ segden, float* __restrict__ Out) {
  int t = blockIdx.x * blockDim.x + threadIdx.x;
  if (t >= (E + N) * 4) return;
  int eid = t >> 2, h = t & 3;
  int s, d;
  if (eid < E) { s = ei[eid]; d = ei[E + eid]; } else { s = d = eid - E; }
  const float* hp = H + (long long)s * 64 + h * 16;
  __builtin_prefetch(hp, 0, 1);               // global_prefetch_b8 on the gather
  float alpha = ebuf[t] / (segden[d * 4 + h] + 1e-16f);
  float* op = Out + (long long)d * 64 + h * 16;
  const float4* hv = (const float4*)hp;       // 64B-aligned head slice
#pragma unroll
  for (int q = 0; q < 4; ++q) {
    float4 v = hv[q];
    atomicAdd(op + 4 * q + 0, v.x * alpha);
    atomicAdd(op + 4 * q + 1, v.y * alpha);
    atomicAdd(op + 4 * q + 2, v.z * alpha);
    atomicAdd(op + 4 * q + 3, v.w * alpha);
  }
}

// ---------------- layer epilogues --------------------------------------------
__global__ void k_finish_concat(const float* __restrict__ Out, const float* __restrict__ bias,
                                float* __restrict__ Nx, int N) {
  int t = blockIdx.x * blockDim.x + threadIdx.x;
  if (t >= N * 64) return;
  float v = Out[t] + bias[t & 63];
  Nx[t] = (v > 0.f) ? v : (__expf(v) - 1.f);  // ELU
}

__global__ void k_finish_mean(const float* __restrict__ Out, const float* __restrict__ bias,
                              float* __restrict__ Nx, int N, int do_elu) {
  int t = blockIdx.x * blockDim.x + threadIdx.x;
  if (t >= N * 16) return;
  int n = t >> 4, c = t & 15;
  const float* op = Out + (long long)n * 64 + c;
  float v = 0.25f * (op[0] + op[16] + op[32] + op[48]) + bias[c];
  if (do_elu) v = (v > 0.f) ? v : (__expf(v) - 1.f);
  Nx[t] = v;
}

// ---------------- global mean pool + MLP head --------------------------------
__global__ void k_pool(const float* __restrict__ H3, const int* __restrict__ batch,
                       float* __restrict__ pooled, float* __restrict__ counts, int N, int B) {
  int t = blockIdx.x * blockDim.x + threadIdx.x;
  if (t >= N * 16) return;
  int n = t >> 4, c = t & 15;
  int g = batch[n];
  if (g < 0) g = 0;
  if (g >= B) g = B - 1;
  atomicAdd(&pooled[g * 16 + c], H3[t]);
  if (c == 0) atomicAdd(&counts[g], 1.0f);
}

__global__ void k_mlp(const float* __restrict__ pooled, const float* __restrict__ counts,
                      const float* __restrict__ stats,
                      const float* __restrict__ fw1, const float* __restrict__ fb1,
                      const float* __restrict__ fw2, const float* __restrict__ fb2,
                      const float* __restrict__ fw3, const float* __restrict__ fb3,
                      float* __restrict__ out, int B) {
  int g = blockIdx.x * blockDim.x + threadIdx.x;
  if (g >= B) return;
  float z[32];
  float cnt = counts[g];
  cnt = (cnt > 1.f) ? cnt : 1.f;
#pragma unroll
  for (int c = 0; c < 16; ++c) z[c] = pooled[g * 16 + c] / cnt;
#pragma unroll
  for (int c = 0; c < 16; ++c) z[16 + c] = stats[g * 16 + c];
  float z1[32];
  for (int j = 0; j < 32; ++j) {
    float s = fb1[j];
    for (int i = 0; i < 32; ++i) s += z[i] * fw1[i * 32 + j];
    z1[j] = (s > 0.f) ? s : 0.f;
  }
  float z2[16];
  for (int j = 0; j < 16; ++j) {
    float s = fb2[j];
    for (int i = 0; i < 32; ++i) s += z1[i] * fw2[i * 16 + j];
    z2[j] = (s > 0.f) ? s : 0.f;
  }
  float o = fb3[0];
  for (int i = 0; i < 16; ++i) o += z2[i] * fw3[i];
  out[g] = o;
}

// ---------------- host driver ------------------------------------------------
static inline int cdiv(long long a, int b) { return (int)((a + b - 1) / b); }

extern "C" void kernel_launch(void* const* d_in, const int* in_sizes, int n_in,
                              void* d_out, int out_size, void* d_ws, size_t ws_size,
                              hipStream_t stream) {
  const float* x     = (const float*)d_in[0];
  const float* stats = (const float*)d_in[1];
  const float* W1  = (const float*)d_in[2];
  const float* a1s = (const float*)d_in[3];
  const float* a1d = (const float*)d_in[4];
  const float* b1  = (const float*)d_in[5];
  const float* W2  = (const float*)d_in[6];
  const float* a2s = (const float*)d_in[7];
  const float* a2d = (const float*)d_in[8];
  const float* b2  = (const float*)d_in[9];
  const float* W3  = (const float*)d_in[10];
  const float* a3s = (const float*)d_in[11];
  const float* a3d = (const float*)d_in[12];
  const float* b3  = (const float*)d_in[13];
  const float* fw1 = (const float*)d_in[14];
  const float* fb1 = (const float*)d_in[15];
  const float* fw2 = (const float*)d_in[16];
  const float* fb2 = (const float*)d_in[17];
  const float* fw3 = (const float*)d_in[18];
  const float* fb3 = (const float*)d_in[19];
  const int* ei    = (const int*)d_in[20];  // [2,E]: row0=src, row1=dst
  const int* batch = (const int*)d_in[21];

  const int N = in_sizes[0] / 16;
  const int E = in_sizes[20] / 2;
  const int B = in_sizes[1] / 16;
  const long long ET = (long long)E + N;    // edges incl. self-loops

  // workspace carving (all offsets 256B-aligned; total ~115 MB -> L2-resident)
  char* w = (char*)d_ws;
  float*    h_feat  = (float*)w;    w += (long long)N * 64 * 4;
  float*    out_acc = (float*)w;    w += (long long)N * 64 * 4;
  float*    next_in = (float*)w;    w += (long long)N * 64 * 4;
  float*    als     = (float*)w;    w += (long long)N * 4 * 4;
  float*    ald     = (float*)w;    w += (long long)N * 4 * 4;
  unsigned* smax    = (unsigned*)w; w += (long long)N * 4 * 4;
  float*    sden    = (float*)w;    w += (long long)N * 4 * 4;
  float*    ebuf    = (float*)w;    w += ET * 4 * 4;
  float*    pooled  = (float*)w;    w += (long long)B * 16 * 4;
  float*    counts  = (float*)w;    w += (long long)B * 4;

  const int T = 256;

  auto layer = [&](const float* in, int K, const float* Wm, const float* avs,
                   const float* avd, const float* bias, int concat, int do_elu) {
    if (K == 64)
      k_gemm_wmma<64><<<cdiv(N, 32), 256, 0, stream>>>(in, Wm, h_feat, N);
    else
      k_gemm_wmma<16><<<cdiv(N, 32), 256, 0, stream>>>(in, Wm, h_feat, N);
    k_alpha<<<cdiv((long long)N * 4, T), T, 0, stream>>>(h_feat, avs, avd, als, ald, N);
    k_fill_u32<<<cdiv((long long)N * 4, T), T, 0, stream>>>(smax, MONO_NEG_INF, (long long)N * 4);
    k_fill_u32<<<cdiv((long long)N * 4, T), T, 0, stream>>>((unsigned*)sden, 0u, (long long)N * 4);
    k_fill_u32<<<cdiv((long long)N * 64, T), T, 0, stream>>>((unsigned*)out_acc, 0u, (long long)N * 64);
    k_edge_max<<<cdiv(ET * 4, T), T, 0, stream>>>(ei, E, N, als, ald, ebuf, smax);
    k_edge_den<<<cdiv(ET * 4, T), T, 0, stream>>>(ei, E, N, smax, ebuf, sden);
    k_edge_aggr<<<cdiv(ET * 4, T), T, 0, stream>>>(ei, E, N, h_feat, ebuf, sden, out_acc);
    if (concat)
      k_finish_concat<<<cdiv((long long)N * 64, T), T, 0, stream>>>(out_acc, bias, next_in, N);
    else
      k_finish_mean<<<cdiv((long long)N * 16, T), T, 0, stream>>>(out_acc, bias, next_in, N, do_elu);
  };

  layer(x,       16, W1, a1s, a1d, b1, /*concat=*/1, /*elu=*/1);  // -> next_in [N,64]
  layer(next_in, 64, W2, a2s, a2d, b2, /*concat=*/0, /*elu=*/1);  // -> next_in [N,16]
  layer(next_in, 16, W3, a3s, a3d, b3, /*concat=*/0, /*elu=*/0);  // -> next_in [N,16]

  k_fill_u32<<<cdiv((long long)B * 16, T), T, 0, stream>>>((unsigned*)pooled, 0u, (long long)B * 16);
  k_fill_u32<<<1, 64, 0, stream>>>((unsigned*)counts, 0u, (long long)B);
  k_pool<<<cdiv((long long)N * 16, T), T, 0, stream>>>(next_in, batch, pooled, counts, N, B);
  k_mlp<<<cdiv(B, 64), 64, 0, stream>>>(pooled, counts, stats, fw1, fb1, fw2, fb2, fw3, fb3,
                                        (float*)d_out, B);
}
